// EnergyAttention_8761733284251
// MI455X (gfx1250) — matrix operand order repaired and checked by reading, not compiled, not run on hardware
//
#include <hip/hip_runtime.h>
#include <hip/hip_bf16.h>

typedef __attribute__((ext_vector_type(16))) _Float16 v16h;
typedef __attribute__((ext_vector_type(8)))  _Float16 v8h;
typedef __attribute__((ext_vector_type(8)))  float    v8f;

#define EMBED 1024
#define HEADS 16
#define QK    64
#define NQ    (HEADS * QK)     // 1024 output cols per projection
#define BATCH 2
#define SEQ   2048
#define ROWS  (BATCH * SEQ)    // 4096
#define SCALE 0.125f           // 1/sqrt(64)
#define NTPW  4                // n-tiles per wave in the score kernel

__global__ void ea_zero_out(float* out) { out[0] = 0.0f; }

__global__ void ea_cvt_f32_f16(const float* __restrict__ in, _Float16* __restrict__ out, int n) {
    int i = blockIdx.x * blockDim.x + threadIdx.x;
    if (i < n) out[i] = (_Float16)in[i];
}

// One wave computes one 16x16 tile of  C = HS(16x1024) * W^T(1024x16), K-loop in steps of 32.
// hs: [4096, 1024] f16 row-major.  w: [1024, 1024] f16, row n_out = (h*64+q), contiguous over d.
// out: [b, h, n, 64] f16, bias[q] added.
__global__ __launch_bounds__(128) void ea_proj_kernel(const _Float16* __restrict__ hs,
                                                      const _Float16* __restrict__ w,
                                                      const float* __restrict__ bias,
                                                      _Float16* __restrict__ out)
{
    const int wave  = (blockIdx.x * blockDim.x + threadIdx.x) >> 5;
    const int lane  = threadIdx.x & 31;
    const int mtile = wave >> 6;    // 0..255  (rows of hidden)
    const int ntile = wave & 63;    // 0..63   (output columns / 16)
    if (mtile >= ROWS / 16) return;

    const int  l16  = lane & 15;
    const bool hi   = lane >= 16;
    const int  aoff = hi ? 16 : 0;   // A lane layout: lo lanes K=0-15; hi lanes K=16-31
    const int  boff = hi ? 16 : 0;   // B lane layout: lane=N col; lo lanes K=0-15; hi lanes K=16-31

    const _Float16* arow = hs + (size_t)(mtile * 16 + l16) * EMBED;
    const _Float16* brow = w  + (size_t)(ntile * 16 + l16) * EMBED;

    v8f c = {};
#pragma unroll 4
    for (int k0 = 0; k0 < EMBED; k0 += 32) {
        v8h a0 = *(const v8h*)(arow + k0 + aoff);
        v8h a1 = *(const v8h*)(arow + k0 + aoff + 8);
        v16h a = __builtin_shufflevector(a0, a1, 0,1,2,3,4,5,6,7,8,9,10,11,12,13,14,15);
        v16h b = *(const v16h*)(brow + k0 + boff);
        c = __builtin_amdgcn_wmma_f32_16x16x32_f16(false, a, false, b, (short)0, c, false, false);
    }

    // C layout: lane group l16 = N, VGPR r -> M = r + (hi ? 8 : 0)
    const int col = ntile * 16 + l16;
    const int h = col >> 6, q = col & 63;
    const float bv = bias[q];
#pragma unroll
    for (int r = 0; r < 8; ++r) {
        const int row = mtile * 16 + r + (hi ? 8 : 0);
        const int bb = row >> 11, n = row & (SEQ - 1);
        out[(((size_t)bb * HEADS + h) * SEQ + n) * QK + q] = (_Float16)(c[r] + bv);
    }
}

// Fused scores + logsumexp. One wave owns a 64-row n-strip (4 resident A tiles) of one
// (b,h) panel and streams all 2048 m in 16-wide tiles: each loaded K tile feeds 8
// v_wmma_f32_16x16x32_f16 (4 independent accumulation chains); exp/accumulate in
// registers; the 536MB scores tensor never touches memory.
__global__ __launch_bounds__(128) void ea_score_lse_kernel(const _Float16* __restrict__ Q,
                                                           const _Float16* __restrict__ K,
                                                           float* __restrict__ out)
{
    const int wave   = (blockIdx.x * blockDim.x + threadIdx.x) >> 5;
    const int lane   = threadIdx.x & 31;
    const int bh     = wave >> 5;      // 0..31
    const int nblk   = wave & 31;      // 0..31  -> rows nblk*64 .. +63
    if (bh >= BATCH * HEADS) return;

    const int  l16  = lane & 15;
    const bool hi   = lane >= 16;
    const int  aoff = hi ? 16 : 0;
    const int  boff = hi ? 16 : 0;

    const _Float16* Qh = Q + (size_t)bh * SEQ * QK;
    const _Float16* Kh = K + (size_t)bh * SEQ * QK;

    // Resident A tiles: 4 x (16 rows of Q), k = 0..63 as two 32-wide halves each.
    v16h a_lo[NTPW], a_hi[NTPW];
#pragma unroll
    for (int t = 0; t < NTPW; ++t) {
        const _Float16* arow = Qh + (size_t)(nblk * 64 + t * 16 + l16) * QK;
        a_lo[t] = __builtin_shufflevector(*(const v8h*)(arow + aoff),
                                          *(const v8h*)(arow + aoff + 8),
                                          0,1,2,3,4,5,6,7,8,9,10,11,12,13,14,15);
        a_hi[t] = __builtin_shufflevector(*(const v8h*)(arow + 32 + aoff),
                                          *(const v8h*)(arow + 32 + aoff + 8),
                                          0,1,2,3,4,5,6,7,8,9,10,11,12,13,14,15);
    }

    float acc[NTPW][8];
#pragma unroll
    for (int t = 0; t < NTPW; ++t)
#pragma unroll
        for (int r = 0; r < 8; ++r) acc[t][r] = 0.0f;

    for (int mt = 0; mt < SEQ / 16; ++mt) {
        const _Float16* brow = Kh + (size_t)(mt * 16 + l16) * QK;
        __builtin_prefetch(brow + 16 * QK, 0, 0);   // global_prefetch_b8 for next m-tile
        v16h b_lo = *(const v16h*)(brow + boff);
        v16h b_hi = *(const v16h*)(brow + boff + 32);
#pragma unroll
        for (int t = 0; t < NTPW; ++t) {
            v8f c = {};
            c = __builtin_amdgcn_wmma_f32_16x16x32_f16(false, a_lo[t], false, b_lo, (short)0, c, false, false);
            c = __builtin_amdgcn_wmma_f32_16x16x32_f16(false, a_hi[t], false, b_hi, (short)0, c, false, false);
#pragma unroll
            for (int r = 0; r < 8; ++r)
                acc[t][r] += __expf(SCALE * c[r]);
        }
    }

    // Row sums: each row M lives across the 16 lanes of a half-wave (N = l16).
#pragma unroll
    for (int off = 1; off < 16; off <<= 1) {
#pragma unroll
        for (int t = 0; t < NTPW; ++t)
#pragma unroll
            for (int r = 0; r < 8; ++r)
                acc[t][r] += __shfl_xor(acc[t][r], off, 32);
    }

    if (l16 == 0) {  // lanes 0 (rows M=0-7 of each tile) and 16 (rows M=8-15)
        float part = 0.0f;
#pragma unroll
        for (int t = 0; t < NTPW; ++t)
#pragma unroll
            for (int r = 0; r < 8; ++r)
                part += __logf(acc[t][r]);
        atomicAdd(out, -part / SCALE);
    }
}

extern "C" void kernel_launch(void* const* d_in, const int* in_sizes, int n_in,
                              void* d_out, int out_size, void* d_ws, size_t ws_size,
                              hipStream_t stream) {
    const float* hidden = (const float*)d_in[0];   // [2, 2048, 1024]
    const float* wq     = (const float*)d_in[1];   // [16, 64, 1024]
    const float* wk     = (const float*)d_in[2];   // [16, 64, 1024]
    const float* qbias  = (const float*)d_in[3];   // [64]
    const float* kbias  = (const float*)d_in[4];   // [64]
    float* out = (float*)d_out;

    // Workspace layout (f16): hs 8MB | wq 2MB | wk 2MB | Q 8MB | K 8MB  (~28MB total)
    _Float16* hs16 = (_Float16*)d_ws;
    _Float16* wq16 = hs16 + (size_t)ROWS * EMBED;
    _Float16* wk16 = wq16 + (size_t)NQ * EMBED;
    _Float16* Q16  = wk16 + (size_t)NQ * EMBED;
    _Float16* K16  = Q16  + (size_t)BATCH * HEADS * SEQ * QK;

    ea_zero_out<<<1, 1, 0, stream>>>(out);

    const int n_hs = ROWS * EMBED;
    const int n_w  = NQ * EMBED;
    ea_cvt_f32_f16<<<(n_hs + 255) / 256, 256, 0, stream>>>(hidden, hs16, n_hs);
    ea_cvt_f32_f16<<<(n_w  + 255) / 256, 256, 0, stream>>>(wq, wq16, n_w);
    ea_cvt_f32_f16<<<(n_w  + 255) / 256, 256, 0, stream>>>(wk, wk16, n_w);

    // 256 m-tiles * 64 n-tiles = 16384 waves, 4 waves/block
    ea_proj_kernel<<<4096, 128, 0, stream>>>(hs16, wq16, qbias, Q16);
    ea_proj_kernel<<<4096, 128, 0, stream>>>(hs16, wk16, kbias, K16);

    // 32 (b,h) * 32 n-blocks(64 rows) = 1024 waves, 4 waves/block
    ea_score_lse_kernel<<<256, 128, 0, stream>>>(Q16, K16, out);
}
